// EGNN_7541962572406
// MI455X (gfx1250) — matrix operand order repaired and checked by reading, not compiled, not run on hardware
//
#include <hip/hip_runtime.h>
#include <hip/hip_bf16.h>

// ---------------------------------------------------------------------------
// EGNN forward for MI455X (gfx1250).  Edge/node MLPs run on
// v_wmma_f32_16x16x32_f16 with 16-row tiles.  Weights are pre-converted to
// f16 AND pre-swizzled into WMMA B-fragment order so each fragment is one
// contiguous 32-byte LDS read per lane (2x ds_load_b128, no shuffles).
// ---------------------------------------------------------------------------

typedef __attribute__((ext_vector_type(16))) _Float16 v16h;
typedef __attribute__((ext_vector_type(8)))  _Float16 v8h;
typedef __attribute__((ext_vector_type(4)))  _Float16 v4h;
typedef __attribute__((ext_vector_type(8)))  float    v8f;

#define TPB 128           // 4 waves per block
#define HID 64
#define K1PAD 160         // 132 -> 160 (5 chunks of 32)

__device__ __forceinline__ float silu_f(float v) {
    return v / (1.0f + __expf(-v));
}

// A fragment: 16x32 f16 tile at column k0 of a row-major [16 x stride] LDS
// tile (stride multiple of 16 halfs).  Lane l: row m=l&15; K halves:
// lanes<16 -> K {0..7,16..23}, lanes>=16 -> K {8..15,24..31}.
__device__ __forceinline__ v16h load_a_frag(const _Float16* base, int stride,
                                            int k0, int lane) {
    int m  = lane & 15;
    int kb = (lane >> 4) * 8;
    const _Float16* p = base + m * stride + k0 + kb;
    v8h lo = *(const v8h*)p;
    v8h hi = *(const v8h*)(p + 16);
    return __builtin_shufflevector(lo, hi, 0, 1, 2, 3, 4, 5, 6, 7,
                                   8, 9, 10, 11, 12, 13, 14, 15);
}

// B fragment from fragment-swizzled weight storage: one contiguous,
// 32B-aligned 16-half run per lane.
__device__ __forceinline__ v16h load_b_frag(const _Float16* Wsw, int kc, int nt,
                                            int lane) {
    return *(const v16h*)(Wsw + (size_t)(((kc * 4 + nt) * 32 + lane) * 16));
}

__device__ __forceinline__ v8f wmma16(v16h a, v16h b, v8f c) {
    return __builtin_amdgcn_wmma_f32_16x16x32_f16(false, a, false, b,
                                                  (short)0, c, false, false);
}

__device__ __forceinline__ void store4h(_Float16* p, float4 v) {
    v4h t;
    t[0] = (_Float16)v.x; t[1] = (_Float16)v.y;
    t[2] = (_Float16)v.z; t[3] = (_Float16)v.w;
    *(v4h*)p = t;
}

// order-preserving float<->uint map for atomicMax-based segment max
__device__ __forceinline__ unsigned f2ord(float f) {
    unsigned u = __float_as_uint(f);
    return (u & 0x80000000u) ? ~u : (u | 0x80000000u);
}
__device__ __forceinline__ float ord2f(unsigned u) {
    return __uint_as_float((u & 0x80000000u) ? (u & 0x7fffffffu) : ~u);
}

// ------------------------------ utility kernels ----------------------------

__global__ void kzero(float* p, int n) {
    int i = blockIdx.x * blockDim.x + threadIdx.x;
    if (i < n) p[i] = 0.0f;
}
__global__ void kcopy(float* d, const float* s, int n) {
    int i = blockIdx.x * blockDim.x + threadIdx.x;
    if (i < n) d[i] = s[i];
}

// f32 [L][srcRows][64] -> f16 B-fragment-swizzled [L][kchunks][4][32][16]
// with zero K-padding.  dst[((kc*4+nt)*32+lane)*16+e] =
//   W[kc*32+(lane>>4)*8+(e&7)+(e>>3)*16][nt*16+(lane&15)]
__global__ void kconvswz(const float* __restrict__ src, _Float16* __restrict__ dst,
                         int layers, int srcRows, int kchunks) {
    int i = blockIdx.x * blockDim.x + threadIdx.x;
    int perLayer = kchunks * 2048;           // 4 nt * 32 lanes * 16 e
    int tot = layers * perLayer;
    if (i >= tot) return;
    int l = i / perLayer;
    int r = i % perLayer;
    int e    = r & 15;
    int lane = (r >> 4) & 31;
    int nt   = (r >> 9) & 3;
    int kc   = r >> 11;
    int k = kc * 32 + (lane >> 4) * 8 + (e & 7) + (e >> 3) * 16;
    int n = nt * 16 + (lane & 15);
    float v = (k < srcRows) ? src[(l * srcRows + k) * HID + n] : 0.0f;
    dst[i] = (_Float16)v;
}

// h = x @ proj_W + proj_b   (N x 26 @ 26 x 64)
__global__ void kproj(const float* __restrict__ x, const float* __restrict__ W,
                      const float* __restrict__ b, float* __restrict__ h, int N) {
    int i = blockIdx.x * blockDim.x + threadIdx.x;
    if (i >= N * HID) return;
    int n = i >> 6, j = i & 63;
    float s = b[j];
#pragma unroll
    for (int k = 0; k < 26; ++k) s += x[n * 26 + k] * W[k * HID + j];
    h[i] = s;
}

// ------------------------------- edge kernel -------------------------------
// Per wave: one 16-edge tile. msg MLP (K=160 padded, K=64), agg scatter,
// optional coord MLP + pos scatter.
__global__ void __launch_bounds__(TPB)
kedge(const float* __restrict__ h, float* __restrict__ agg,
      const float* __restrict__ pos, float* __restrict__ posn,
      const int* __restrict__ ei, const float* __restrict__ ea,
      const _Float16* __restrict__ W1, const _Float16* __restrict__ W2,
      const _Float16* __restrict__ CW1, const float* __restrict__ CW2,
      const float* __restrict__ b1, const float* __restrict__ b2,
      const float* __restrict__ cb1, const float* __restrict__ cb2p,
      int hasCoord, int E) {
    __shared__ alignas(32) _Float16 sW1[K1PAD * HID];   // 20480 B (swizzled)
    __shared__ alignas(32) _Float16 sW2[HID * HID];     //  8192 B (swizzled)
    __shared__ alignas(32) float    sCW2[HID];
    __shared__ alignas(32) int      sRow[4][16];
    __shared__ alignas(32) float    sDiff[4][16][3];
    __shared__ alignas(32) _Float16 sH[4][16 * HID];    //  8192 B
    __shared__ alignas(32) _Float16 sA[4][16 * K1PAD];  // 20480 B (reused as CW1)

    const int tid = threadIdx.x, lane = tid & 31, w = tid >> 5;

    // stage weights with 16B vector copies
    {
        const uint4* s1 = (const uint4*)W1;
        uint4* d1 = (uint4*)sW1;
        for (int i = tid; i < K1PAD * HID / 8; i += TPB) d1[i] = s1[i];
        const uint4* s2 = (const uint4*)W2;
        uint4* d2 = (uint4*)sW2;
        for (int i = tid; i < HID * HID / 8; i += TPB) d2[i] = s2[i];
        if (hasCoord)
            for (int i = tid; i < HID; i += TPB) sCW2[i] = CW2[i];
    }

    // ---- stage A tile: [h[row] | h[col] | dist | edge_attr | 0-pad] -------
    const int et   = (blockIdx.x * 4 + w) * 16;
    const int row  = lane & 15;
    const int half = lane >> 4;
    const int e    = et + row;
    _Float16* A = sA[w];
    _Float16* Ar = A + row * K1PAD;
    if (e < E) {
        int ridx = ei[e], cidx = ei[E + e];
        float dx = pos[ridx * 3 + 0] - pos[cidx * 3 + 0];
        float dy = pos[ridx * 3 + 1] - pos[cidx * 3 + 1];
        float dz = pos[ridx * 3 + 2] - pos[cidx * 3 + 2];
        float dist = sqrtf(dx * dx + dy * dy + dz * dz);
        if (half == 0) {
            sRow[w][row] = ridx;
            sDiff[w][row][0] = dx; sDiff[w][row][1] = dy; sDiff[w][row][2] = dz;
            const float4* src = (const float4*)(h + (size_t)ridx * HID);
#pragma unroll
            for (int q = 0; q < 16; ++q) store4h(Ar + q * 4, src[q]);
        } else {
            const float4* src = (const float4*)(h + (size_t)cidx * HID);
#pragma unroll
            for (int q = 0; q < 16; ++q) store4h(Ar + 64 + q * 4, src[q]);
            v4h t;
            t[0] = (_Float16)dist;
            t[1] = (_Float16)ea[e * 3 + 0];
            t[2] = (_Float16)ea[e * 3 + 1];
            t[3] = (_Float16)ea[e * 3 + 2];
            *(v4h*)(Ar + 128) = t;
            v4h z = {};
#pragma unroll
            for (int q = 0; q < 7; ++q) *(v4h*)(Ar + 132 + q * 4) = z;
        }
    } else {
        if (half == 0) sRow[w][row] = -1;
        v8h z = {};
#pragma unroll
        for (int q = 0; q < 10; ++q) *(v8h*)(Ar + half * 80 + q * 8) = z;
    }
    __syncthreads();

    const int ncol = lane & 15;
    const int mb   = (lane >> 4) * 8;

    // ---- GEMM1: [16x160] @ [160x64] -> silu -> sH -------------------------
    {
        v8f acc[4];
#pragma unroll
        for (int nt = 0; nt < 4; ++nt) acc[nt] = v8f{};
#pragma unroll
        for (int kc = 0; kc < 5; ++kc) {
            v16h a = load_a_frag(A, K1PAD, kc * 32, lane);
#pragma unroll
            for (int nt = 0; nt < 4; ++nt)
                acc[nt] = wmma16(a, load_b_frag(sW1, kc, nt, lane), acc[nt]);
        }
#pragma unroll
        for (int nt = 0; nt < 4; ++nt) {
            int n = nt * 16 + ncol;
            float bb = b1[n];
#pragma unroll
            for (int r = 0; r < 8; ++r)
                sH[w][(mb + r) * HID + n] = (_Float16)silu_f(acc[nt][r] + bb);
        }
    }
    __syncthreads();

    // ---- GEMM2: [16x64] @ [64x64] -> silu -> msg (regs) -------------------
    float msgv[4][8];
    {
        v8f acc[4];
#pragma unroll
        for (int nt = 0; nt < 4; ++nt) acc[nt] = v8f{};
#pragma unroll
        for (int kc = 0; kc < 2; ++kc) {
            v16h a = load_a_frag(sH[w], HID, kc * 32, lane);
#pragma unroll
            for (int nt = 0; nt < 4; ++nt)
                acc[nt] = wmma16(a, load_b_frag(sW2, kc, nt, lane), acc[nt]);
        }
#pragma unroll
        for (int nt = 0; nt < 4; ++nt) {
            int n = nt * 16 + ncol;
            float bb = b2[n];
#pragma unroll
            for (int r = 0; r < 8; ++r) msgv[nt][r] = silu_f(acc[nt][r] + bb);
        }
    }

    // ---- scatter agg[row] += msg ------------------------------------------
#pragma unroll
    for (int nt = 0; nt < 4; ++nt) {
        int n = nt * 16 + ncol;
#pragma unroll
        for (int r = 0; r < 8; ++r) {
            int rr = sRow[w][mb + r];
            if (rr >= 0) atomicAdd(&agg[(size_t)rr * HID + n], msgv[nt][r]);
        }
    }

    // ---- coord update (layers 0..2) ---------------------------------------
    if (hasCoord) {
        __syncthreads();
        // write msg as f16 into own sH tile; stage coord_W1 over the sA space
#pragma unroll
        for (int nt = 0; nt < 4; ++nt)
#pragma unroll
            for (int r = 0; r < 8; ++r)
                sH[w][(mb + r) * HID + nt * 16 + ncol] = (_Float16)msgv[nt][r];
        _Float16* sCW1 = &sA[0][0];
        {
            const uint4* sc = (const uint4*)CW1;
            uint4* dc = (uint4*)sCW1;
            for (int i = tid; i < HID * HID / 8; i += TPB) dc[i] = sc[i];
        }
        __syncthreads();

        float part[8] = {0.f, 0.f, 0.f, 0.f, 0.f, 0.f, 0.f, 0.f};
#pragma unroll
        for (int nt = 0; nt < 4; ++nt) {
            v8f acc = v8f{};
#pragma unroll
            for (int kc = 0; kc < 2; ++kc) {
                v16h a = load_a_frag(sH[w], HID, kc * 32, lane);
                acc = wmma16(a, load_b_frag(sCW1, kc, nt, lane), acc);
            }
            int n = nt * 16 + ncol;
            float cb = cb1[n];
            float cw = sCW2[n];
#pragma unroll
            for (int r = 0; r < 8; ++r) part[r] += silu_f(acc[r] + cb) * cw;
        }
        // reduce over the 16 lanes of each half (xor masks stay in-group)
#pragma unroll
        for (int off = 8; off > 0; off >>= 1)
#pragma unroll
            for (int r = 0; r < 8; ++r) part[r] += __shfl_xor(part[r], off);

        if (ncol == 0) {                      // lanes 0 and 16
            float cb2v = cb2p[0];
#pragma unroll
            for (int r = 0; r < 8; ++r) {
                int m  = mb + r;
                int rr = sRow[w][m];
                if (rr >= 0) {
                    float we = part[r] + cb2v;
                    atomicAdd(&posn[rr * 3 + 0], sDiff[w][m][0] * we);
                    atomicAdd(&posn[rr * 3 + 1], sDiff[w][m][1] * we);
                    atomicAdd(&posn[rr * 3 + 2], sDiff[w][m][2] * we);
                }
            }
        }
    }
}

// ------------------------------- node kernel -------------------------------
// h += silu([h|agg] @ W1 + b1) @ W2 + b2   per 16-node wave tile
__global__ void __launch_bounds__(TPB)
knode(float* __restrict__ h, const float* __restrict__ agg,
      const _Float16* __restrict__ W1, const _Float16* __restrict__ W2,
      const float* __restrict__ b1, const float* __restrict__ b2, int N) {
    __shared__ alignas(32) _Float16 sW1[128 * HID];     // 16384 B (swizzled)
    __shared__ alignas(32) _Float16 sW2[HID * HID];     //  8192 B (swizzled)
    __shared__ alignas(32) _Float16 sA[4][16 * 128];    // 16384 B
    __shared__ alignas(32) _Float16 sH[4][16 * HID];    //  8192 B

    const int tid = threadIdx.x, lane = tid & 31, w = tid >> 5;
    {
        const uint4* s1 = (const uint4*)W1;
        uint4* d1 = (uint4*)sW1;
        for (int i = tid; i < 128 * HID / 8; i += TPB) d1[i] = s1[i];
        const uint4* s2 = (const uint4*)W2;
        uint4* d2 = (uint4*)sW2;
        for (int i = tid; i < HID * HID / 8; i += TPB) d2[i] = s2[i];
    }

    const int nb   = (blockIdx.x * 4 + w) * 16;
    const int row  = lane & 15;
    const int half = lane >> 4;
    const int node = nb + row;
    _Float16* A = sA[w];
    _Float16* Ar = A + row * 128 + half * HID;
    if (node < N) {
        const float4* src = (const float4*)((half ? agg : h) + (size_t)node * HID);
#pragma unroll
        for (int q = 0; q < 16; ++q) store4h(Ar + q * 4, src[q]);
    } else {
        v8h z = {};
#pragma unroll
        for (int q = 0; q < 8; ++q) *(v8h*)(Ar + q * 8) = z;
    }
    __syncthreads();

    const int ncol = lane & 15;
    const int mb   = (lane >> 4) * 8;

    v8f acc[4];
#pragma unroll
    for (int nt = 0; nt < 4; ++nt) acc[nt] = v8f{};
#pragma unroll
    for (int kc = 0; kc < 4; ++kc) {
        v16h a = load_a_frag(A, 128, kc * 32, lane);
#pragma unroll
        for (int nt = 0; nt < 4; ++nt)
            acc[nt] = wmma16(a, load_b_frag(sW1, kc, nt, lane), acc[nt]);
    }
#pragma unroll
    for (int nt = 0; nt < 4; ++nt) {
        int n = nt * 16 + ncol;
        float bb = b1[n];
#pragma unroll
        for (int r = 0; r < 8; ++r)
            sH[w][(mb + r) * HID + n] = (_Float16)silu_f(acc[nt][r] + bb);
    }
    __syncthreads();

#pragma unroll
    for (int nt = 0; nt < 4; ++nt) acc[nt] = v8f{};
#pragma unroll
    for (int kc = 0; kc < 2; ++kc) {
        v16h a = load_a_frag(sH[w], HID, kc * 32, lane);
#pragma unroll
        for (int nt = 0; nt < 4; ++nt)
            acc[nt] = wmma16(a, load_b_frag(sW2, kc, nt, lane), acc[nt]);
    }
#pragma unroll
    for (int nt = 0; nt < 4; ++nt) {
        int n = nt * 16 + ncol;
        float bb = b2[n];
#pragma unroll
        for (int r = 0; r < 8; ++r) {
            int nd = nb + mb + r;
            if (nd < N) {
                size_t gi = (size_t)nd * HID + n;
                h[gi] = h[gi] + (acc[nt][r] + bb);
            }
        }
    }
}

// ------------------------------- pooling / cls -----------------------------

__global__ void kpool(const float* __restrict__ h, const int* __restrict__ batch,
                      float* __restrict__ sums, float* __restrict__ counts,
                      unsigned* __restrict__ maxb, int N) {
    int n = blockIdx.x * blockDim.x + threadIdx.x;
    if (n >= N) return;
    int g = batch[n];
    atomicAdd(&counts[g], 1.0f);
    for (int j = 0; j < HID; ++j) {
        float v = h[(size_t)n * HID + j];
        atomicAdd(&sums[g * HID + j], v);
        atomicMax(&maxb[g * HID + j], f2ord(v));
    }
}

__global__ void kcls(const float* __restrict__ sums, const float* __restrict__ counts,
                     const unsigned* __restrict__ maxb,
                     const float* __restrict__ W1, const float* __restrict__ b1,
                     const float* __restrict__ W2, const float* __restrict__ b2,
                     float* __restrict__ out, int G) {
    int g = blockIdx.x * blockDim.x + threadIdx.x;
    if (g >= G) return;
    float pooled[128];
    float c = counts[g]; if (c < 1.0f) c = 1.0f;
    for (int j = 0; j < HID; ++j) pooled[j] = sums[g * HID + j] / c;
    for (int j = 0; j < HID; ++j) pooled[HID + j] = ord2f(maxb[g * HID + j]);
    float hid[HID];
    for (int j = 0; j < HID; ++j) {
        float s = b1[j];
        for (int k = 0; k < 128; ++k) s += pooled[k] * W1[k * HID + j];
        hid[j] = fmaxf(s, 0.0f);
    }
    for (int cc = 0; cc < 2; ++cc) {
        float s = b2[cc];
        for (int j = 0; j < HID; ++j) s += hid[j] * W2[j * 2 + cc];
        out[g * 2 + cc] = s;
    }
}

// ------------------------------- host driver -------------------------------

extern "C" void kernel_launch(void* const* d_in, const int* in_sizes, int n_in,
                              void* d_out, int out_size, void* d_ws, size_t ws_size,
                              hipStream_t stream) {
    const float* x       = (const float*)d_in[0];
    const float* pos     = (const float*)d_in[1];
    const int*   ei      = (const int*)  d_in[2];
    const float* ea      = (const float*)d_in[3];
    const int*   batch   = (const int*)  d_in[4];
    const float* projW   = (const float*)d_in[5];
    const float* projb   = (const float*)d_in[6];
    const float* msgW1   = (const float*)d_in[7];
    const float* msgb1   = (const float*)d_in[8];
    const float* msgW2   = (const float*)d_in[9];
    const float* msgb2   = (const float*)d_in[10];
    const float* nodeW1  = (const float*)d_in[11];
    const float* nodeb1  = (const float*)d_in[12];
    const float* nodeW2  = (const float*)d_in[13];
    const float* nodeb2  = (const float*)d_in[14];
    const float* coordW1 = (const float*)d_in[15];
    const float* coordb1 = (const float*)d_in[16];
    const float* coordW2 = (const float*)d_in[17];
    const float* coordb2 = (const float*)d_in[18];
    const float* clsW1   = (const float*)d_in[19];
    const float* clsb1   = (const float*)d_in[20];
    const float* clsW2   = (const float*)d_in[21];
    const float* clsb2   = (const float*)d_in[22];

    const int N = in_sizes[4];
    const int E = in_sizes[2] / 2;
    const int G = 128;
    const int L = 4;

    char* ws = (char*)d_ws;
    size_t cur = 0;
    auto alloc = [&](size_t bytes) -> char* {
        char* p = ws + cur;
        cur = (cur + bytes + 255) & ~(size_t)255;
        return p;
    };
    float* h     = (float*)alloc((size_t)N * HID * 4);
    float* agg   = (float*)alloc((size_t)N * HID * 4);
    float* posA  = (float*)alloc((size_t)N * 3 * 4);
    float* posB  = (float*)alloc((size_t)N * 3 * 4);
    float* poolb = (float*)alloc((size_t)(G * HID + G + G * HID) * 4);
    float*    sums   = poolb;
    float*    counts = poolb + G * HID;
    unsigned* maxb   = (unsigned*)(poolb + G * HID + G);
    _Float16* msgW1h  = (_Float16*)alloc((size_t)L * K1PAD * HID * 2);
    _Float16* msgW2h  = (_Float16*)alloc((size_t)L * HID * HID * 2);
    _Float16* nodeW1h = (_Float16*)alloc((size_t)L * 128 * HID * 2);
    _Float16* nodeW2h = (_Float16*)alloc((size_t)L * HID * HID * 2);
    _Float16* crdW1h  = (_Float16*)alloc((size_t)(L - 1) * HID * HID * 2);

    auto g1 = [](int n) { return (n + 255) / 256; };

    // weight conversion: f32 -> f16, K-padded, B-fragment swizzled
    kconvswz<<<g1(L * 5 * 2048), 256, 0, stream>>>(msgW1, msgW1h, L, 132, 5);
    kconvswz<<<g1(L * 2 * 2048), 256, 0, stream>>>(msgW2, msgW2h, L, HID, 2);
    kconvswz<<<g1(L * 4 * 2048), 256, 0, stream>>>(nodeW1, nodeW1h, L, 128, 4);
    kconvswz<<<g1(L * 2 * 2048), 256, 0, stream>>>(nodeW2, nodeW2h, L, HID, 2);
    kconvswz<<<g1((L - 1) * 2 * 2048), 256, 0, stream>>>(coordW1, crdW1h, L - 1, HID, 2);

    // h = x @ proj_W + b ; pos working copy
    kproj<<<g1(N * HID), 256, 0, stream>>>(x, projW, projb, h, N);
    kcopy<<<g1(N * 3), 256, 0, stream>>>(posA, pos, N * 3);

    float* pc = posA;
    float* pn = posB;
    const int eBlocks = (E + 63) / 64;
    const int nBlocks = (N + 63) / 64;

    for (int i = 0; i < L; ++i) {
        const int hc = (i < L - 1) ? 1 : 0;
        kzero<<<g1(N * HID), 256, 0, stream>>>(agg, N * HID);
        if (hc) kcopy<<<g1(N * 3), 256, 0, stream>>>(pn, pc, N * 3);
        kedge<<<eBlocks, TPB, 0, stream>>>(
            h, agg, pc, hc ? pn : posB, ei, ea,
            msgW1h + (size_t)i * K1PAD * HID,
            msgW2h + (size_t)i * HID * HID,
            hc ? (crdW1h + (size_t)i * HID * HID) : crdW1h,
            hc ? (coordW2 + i * HID) : coordW2,
            msgb1 + i * HID, msgb2 + i * HID,
            hc ? (coordb1 + i * HID) : coordb1,
            hc ? (coordb2 + i) : coordb2,
            hc, E);
        knode<<<nBlocks, TPB, 0, stream>>>(
            h, agg,
            nodeW1h + (size_t)i * 128 * HID,
            nodeW2h + (size_t)i * HID * HID,
            nodeb1 + i * HID, nodeb2 + i * HID, N);
        if (hc) { float* t = pc; pc = pn; pn = t; }
    }

    // pooling + classifier
    kzero<<<g1(G * HID + G + G * HID), 256, 0, stream>>>(poolb, G * HID + G + G * HID);
    kpool<<<g1(N), 256, 0, stream>>>(h, batch, sums, counts, maxb, N);
    kcls<<<1, 128, 0, stream>>>(sums, counts, maxb, clsW1, clsb1, clsW2, clsb2,
                                (float*)d_out, G);
}
